// MultiHeadAttention_71038759076084
// MI455X (gfx1250) — matrix operand order repaired and checked by reading, not compiled, not run on hardware
//
#include <hip/hip_runtime.h>

// Problem constants (match reference)
#define BB   2
#define SS   2048
#define DD   1024
#define HH   16
#define DKK  64
#define MM   (BB*SS)    // 4096 rows total
#define HDK  (HH*DKK)   // 1024

typedef __attribute__((ext_vector_type(16))) __bf16 v16bf;
typedef __attribute__((ext_vector_type(8)))  __bf16 v8bf;
typedef __attribute__((ext_vector_type(8)))  float  v8f;

typedef __attribute__((address_space(3))) __bf16 lds_bf16;

// ---- float -> bf16 (RNE) via bit manipulation (storage-only use of __bf16) ----
static __device__ __forceinline__ __bf16 f2bf(float f) {
    union { float f; unsigned u; } x; x.f = f;
    unsigned r = x.u + 0x7fffu + ((x.u >> 16) & 1u);
    unsigned short h = (unsigned short)(r >> 16);
    return *reinterpret_cast<__bf16*>(&h);
}

// ---- WMMA A-operand fragment: 16x32 bf16, ISA 7.12.2 interleaved K layout ----
// lane<16 : M=lane,     elems 0..7 -> K=k0+0..7,   elems 8..15 -> K=k0+16..23
// lane>=16: M=lane-16,  elems 0..7 -> K=k0+8..15,  elems 8..15 -> K=k0+24..31
static __device__ __forceinline__ v16bf load_fragA(const __bf16* base, int stride,
                                                   int row0, int k0, int lane) {
    const int lr = lane & 15;
    const int hi = lane >> 4;
    const __bf16* p = base + (size_t)(row0 + lr) * (size_t)stride + k0 + hi * 8;
    v8bf lo = *reinterpret_cast<const v8bf*>(p);
    v8bf up = *reinterpret_cast<const v8bf*>(p + 16);
    v16bf r;
#pragma unroll
    for (int i = 0; i < 8; ++i) { r[i] = lo[i]; r[i + 8] = up[i]; }
    return r;
}

// ---- WMMA B-operand fragment: 32x16 bf16 ----
// lane<16 : N=lane,     elems 0..7 -> K=k0+0..7,   elems 8..15 -> K=k0+16..23
// lane>=16: N=lane-16,  elems 0..7 -> K=k0+8..15,  elems 8..15 -> K=k0+24..31
// Source stored transposed: row n0+lane holds contiguous K.
static __device__ __forceinline__ v16bf load_fragB(const __bf16* base, int stride,
                                                   int n0, int k0, int lane) {
    const int lr = lane & 15;
    const int hi = lane >> 4;
    const __bf16* p = base + (size_t)(n0 + lr) * (size_t)stride + k0 + hi * 8;
    v8bf lo = *reinterpret_cast<const v8bf*>(p);
    v8bf up = *reinterpret_cast<const v8bf*>(p + 16);
    v16bf r;
#pragma unroll
    for (int i = 0; i < 8; ++i) { r[i] = lo[i]; r[i + 8] = up[i]; }
    return r;
}

static __device__ __forceinline__ v8f wmma_bf16(v16bf a, v16bf b, v8f c) {
    return __builtin_amdgcn_wmma_f32_16x16x32_bf16(false, a, false, b,
                                                   (short)0, c, false, false);
}

// ---- async global -> LDS staging of one 64x32 bf16 B tile (4 KB) ----
// Bt row stride is 1024 elems for both weight layouts used here.
// 128 threads: each issues two b128 async copies (16 B each lane).
static __device__ __forceinline__ void stage_B_async(const __bf16* Bt, int d,
                                                     __bf16* ldsB, int tid) {
    const int seg  = tid & 3;        // 4 x 16B segments per 64B row
    const int row0 = tid >> 2;       // 32 rows per pass
#pragma unroll
    for (int rr = 0; rr < 2; ++rr) {
        const int row = row0 + rr * 32;
        const __bf16* g = Bt + (size_t)row * 1024 + d + seg * 8;
        unsigned long long ga = (unsigned long long)(size_t)g;
        unsigned la = (unsigned)(size_t)(lds_bf16*)(ldsB + row * 32 + seg * 8);
        asm volatile("global_load_async_to_lds_b128 %0, %1, off"
                     :: "v"(la), "v"(ga) : "memory");
    }
}

static __device__ __forceinline__ void wait_async_all() {
    asm volatile("s_wait_asynccnt 0x0" ::: "memory");
}

// ===================== pack / convert kernels =====================

__global__ void k_cvt(const float* __restrict__ in, __bf16* __restrict__ out, int n) {
    int i = blockIdx.x * blockDim.x + threadIdx.x;
    int st = gridDim.x * blockDim.x;
    for (; i < n; i += st) out[i] = f2bf(in[i]);
}

// W: [H, D, DK] (f32) -> Wt: [H, DK, D] (bf16)   (B-operand wants column-contig K)
__global__ void k_twt(const float* __restrict__ W, __bf16* __restrict__ Wt) {
    const int n = HH * DD * DKK;
    int i = blockIdx.x * blockDim.x + threadIdx.x;
    int st = gridDim.x * blockDim.x;
    for (; i < n; i += st) {
        int h = i / (DD * DKK);
        int r = i - h * (DD * DKK);
        int d = r / DKK;
        int k = r - d * DKK;
        Wt[((size_t)h * DKK + k) * DD + d] = f2bf(W[i]);
    }
}

// Wo: [HDK, D] (f32) -> WoT: [D, HDK] (bf16)
__global__ void k_two(const float* __restrict__ W, __bf16* __restrict__ Wt) {
    const int n = HDK * DD;
    int i = blockIdx.x * blockDim.x + threadIdx.x;
    int st = gridDim.x * blockDim.x;
    for (; i < n; i += st) {
        int k = i / DD;
        int d = i - k * DD;
        Wt[(size_t)d * HDK + k] = f2bf(W[i]);
    }
}

// ===================== per-head QKV projection =====================
// Out = X[4096,1024] * Wh[1024,64] per head. Each wave: 64x64 tile.
// B tile (shared by all 4 waves) staged via async global->LDS, double buffered.
// storeT=0 -> [B,H,S,DK] (Q,K); storeT=1 -> [B,H,DK,S] (V transposed).
__global__ __launch_bounds__(128) void k_proj(const __bf16* __restrict__ X,
                                              const __bf16* __restrict__ Wt,
                                              __bf16* __restrict__ Out,
                                              int storeT) {
    __shared__ __bf16 ldsB[2][64][32];
    const int tid  = threadIdx.x;
    const int lane = threadIdx.x & 31;
    const int wid  = threadIdx.x >> 5;
    const int h  = blockIdx.y;
    const int m0 = (blockIdx.x * 4 + wid) * 64;
    const __bf16* Wh = Wt + (size_t)h * DKK * DD;

    v8f zero = {};
    v8f acc[4][4];
#pragma unroll
    for (int i = 0; i < 4; ++i)
#pragma unroll
        for (int j = 0; j < 4; ++j) acc[i][j] = zero;

    stage_B_async(Wh, 0, &ldsB[0][0][0], tid);

    for (int d = 0; d < DD; d += 32) {
        const int cur = (d >> 5) & 1;
        wait_async_all();        // this wave's portion of current tile landed
        __syncthreads();         // whole tile visible to all waves
        if (d + 32 < DD)
            stage_B_async(Wh, d + 32, &ldsB[cur ^ 1][0][0], tid);

        v16bf a[4], b[4];
#pragma unroll
        for (int i = 0; i < 4; ++i) a[i] = load_fragA(X, DD, m0 + 16 * i, d, lane);
#pragma unroll
        for (int j = 0; j < 4; ++j) b[j] = load_fragB(&ldsB[cur][0][0], 32, 16 * j, 0, lane);
        if (d + 128 < DD)
            __builtin_prefetch(X + (size_t)(m0 + (lane & 15)) * DD + d + 128, 0, 1);
#pragma unroll
        for (int i = 0; i < 4; ++i)
#pragma unroll
            for (int j = 0; j < 4; ++j)
                acc[i][j] = wmma_bf16(a[i], b[j], acc[i][j]);
    }

    const int lr = lane & 15, hi2 = lane >> 4;
#pragma unroll
    for (int i = 0; i < 4; ++i) {
#pragma unroll
        for (int j = 0; j < 4; ++j) {
            const int dk = 16 * j + lr;
#pragma unroll
            for (int r = 0; r < 8; ++r) {
                const int m  = m0 + 16 * i + r + 8 * hi2;
                const int b_ = m >> 11;          // m / S
                const int s  = m & (SS - 1);
                const __bf16 val = f2bf(acc[i][j][r]);
                if (!storeT)
                    Out[(((size_t)b_ * HH + h) * SS + s) * DKK + dk] = val;
                else
                    Out[(((size_t)b_ * HH + h) * DKK + dk) * SS + s] = val;
            }
        }
    }
}

// ===================== flash attention =====================
// One wave = one (b,h, 16-row q tile). Online softmax over key chunks of 32.
__global__ __launch_bounds__(128) void k_attn(const __bf16* __restrict__ Qh,  // [B,H,S,DK]
                                              const __bf16* __restrict__ Kh,  // [B,H,S,DK]
                                              const __bf16* __restrict__ VhT, // [B,H,DK,S]
                                              __bf16* __restrict__ Oh) {      // [B,S,H,DK]
    __shared__ __bf16 Pst[4][16][32];
    const int lane = threadIdx.x & 31;
    const int wid  = threadIdx.x >> 5;
    const int lr = lane & 15, hi2 = lane >> 4;
    const int flat = blockIdx.x * 4 + wid;           // [0, 4096)
    const int qt = flat & 127;                       // S/16 tiles
    const int h  = (flat >> 7) & (HH - 1);
    const int b  = flat >> 11;
    const size_t bh = (size_t)b * HH + h;
    const __bf16* Qb = Qh  + bh * (size_t)SS * DKK;
    const __bf16* Kb = Kh  + bh * (size_t)SS * DKK;
    const __bf16* Vb = VhT + bh * (size_t)DKK * SS;
    const int q0 = qt * 16;

    v16bf aQ0 = load_fragA(Qb, DKK, q0, 0,  lane);
    v16bf aQ1 = load_fragA(Qb, DKK, q0, 32, lane);

    v8f zero = {};
    v8f o0 = zero, o1 = zero, o2 = zero, o3 = zero;
    float rowmax[8], rowsum[8];
#pragma unroll
    for (int r = 0; r < 8; ++r) { rowmax[r] = -3.0e38f; rowsum[r] = 0.0f; }

    for (int k0 = 0; k0 < SS; k0 += 32) {
        // scores: two 16x16 tiles, contraction over DK=64 (two 32-chunks)
        v16bf bk00 = load_fragB(Kb, DKK, k0,      0,  lane);
        v16bf bk01 = load_fragB(Kb, DKK, k0,      32, lane);
        v16bf bk10 = load_fragB(Kb, DKK, k0 + 16, 0,  lane);
        v16bf bk11 = load_fragB(Kb, DKK, k0 + 16, 32, lane);
        v8f c0 = zero, c1 = zero;
        c0 = wmma_bf16(aQ0, bk00, c0);
        c0 = wmma_bf16(aQ1, bk01, c0);
        c1 = wmma_bf16(aQ0, bk10, c1);
        c1 = wmma_bf16(aQ1, bk11, c1);

        // online softmax; C-layout: lane half hi2 owns rows r + 8*hi2, col = lr (+16)
#pragma unroll
        for (int r = 0; r < 8; ++r) {
            float s0 = c0[r] * 0.125f;   // 1/sqrt(64)
            float s1 = c1[r] * 0.125f;
            float mn = fmaxf(s0, s1);
#pragma unroll
            for (int mk = 1; mk < 16; mk <<= 1)
                mn = fmaxf(mn, __shfl_xor(mn, mk, 32));
            mn = fmaxf(mn, rowmax[r]);
            const float corr = __expf(rowmax[r] - mn);
            rowmax[r] = mn;
            const float p0 = __expf(s0 - mn);
            const float p1 = __expf(s1 - mn);
            float ps = p0 + p1;
#pragma unroll
            for (int mk = 1; mk < 16; mk <<= 1)
                ps += __shfl_xor(ps, mk, 32);
            rowsum[r] = rowsum[r] * corr + ps;
            o0[r] *= corr; o1[r] *= corr; o2[r] *= corr; o3[r] *= corr;
            Pst[wid][r + 8 * hi2][lr]      = f2bf(p0);
            Pst[wid][r + 8 * hi2][lr + 16] = f2bf(p1);
        }
        __syncthreads();

        // P(16x32) * V(32x64): A from LDS (layout shuffle), B contiguous from VhT
        v16bf aP  = load_fragA(&Pst[wid][0][0], 32, 0, 0, lane);
        v16bf bv0 = load_fragB(Vb, SS, 0,  k0, lane);
        v16bf bv1 = load_fragB(Vb, SS, 16, k0, lane);
        v16bf bv2 = load_fragB(Vb, SS, 32, k0, lane);
        v16bf bv3 = load_fragB(Vb, SS, 48, k0, lane);
        o0 = wmma_bf16(aP, bv0, o0);
        o1 = wmma_bf16(aP, bv1, o1);
        o2 = wmma_bf16(aP, bv2, o2);
        o3 = wmma_bf16(aP, bv3, o3);
        __syncthreads();
    }

    // normalize + store concat layout [B,S,H,DK]
#pragma unroll
    for (int r = 0; r < 8; ++r) {
        const float inv = 1.0f / rowsum[r];
        const int s = q0 + r + 8 * hi2;
        const size_t rowbase = (((size_t)b * SS + s) * HH + h) * DKK;
        Oh[rowbase +  0 + lr] = f2bf(o0[r] * inv);
        Oh[rowbase + 16 + lr] = f2bf(o1[r] * inv);
        Oh[rowbase + 32 + lr] = f2bf(o2[r] * inv);
        Oh[rowbase + 48 + lr] = f2bf(o3[r] * inv);
    }
}

// ===================== output projection =====================
// out[4096,1024] = Oh[4096,1024] * Wo[1024,1024] + bo; f32 result.
// B tile (WoT rows n0..n0+63, shared by all 4 waves) staged async, double buffered.
__global__ __launch_bounds__(128) void k_oproj(const __bf16* __restrict__ A,    // [M, HDK]
                                               const __bf16* __restrict__ WoT,  // [D, HDK]
                                               const float* __restrict__ bo,
                                               float* __restrict__ out) {       // [M, D]
    __shared__ __bf16 ldsB[2][64][32];
    const int tid  = threadIdx.x;
    const int lane = threadIdx.x & 31;
    const int wid  = threadIdx.x >> 5;
    const int m0 = (blockIdx.x * 4 + wid) * 64;
    const int n0 = blockIdx.y * 64;
    const __bf16* Bt = WoT + (size_t)n0 * HDK;   // 64 rows of WoT

    v8f zero = {};
    v8f acc[4][4];
#pragma unroll
    for (int i = 0; i < 4; ++i)
#pragma unroll
        for (int j = 0; j < 4; ++j) acc[i][j] = zero;

    stage_B_async(Bt, 0, &ldsB[0][0][0], tid);

    for (int kk = 0; kk < HDK; kk += 32) {
        const int cur = (kk >> 5) & 1;
        wait_async_all();
        __syncthreads();
        if (kk + 32 < HDK)
            stage_B_async(Bt, kk + 32, &ldsB[cur ^ 1][0][0], tid);

        v16bf a[4], b[4];
#pragma unroll
        for (int i = 0; i < 4; ++i) a[i] = load_fragA(A, HDK, m0 + 16 * i, kk, lane);
#pragma unroll
        for (int j = 0; j < 4; ++j) b[j] = load_fragB(&ldsB[cur][0][0], 32, 16 * j, 0, lane);
        if (kk + 128 < HDK)
            __builtin_prefetch(A + (size_t)(m0 + (lane & 15)) * HDK + kk + 128, 0, 1);
#pragma unroll
        for (int i = 0; i < 4; ++i)
#pragma unroll
            for (int j = 0; j < 4; ++j)
                acc[i][j] = wmma_bf16(a[i], b[j], acc[i][j]);
    }

    const int lr = lane & 15, hi2 = lane >> 4;
#pragma unroll
    for (int i = 0; i < 4; ++i) {
#pragma unroll
        for (int j = 0; j < 4; ++j) {
            const int n = n0 + 16 * j + lr;
            const float bias = bo[n];
#pragma unroll
            for (int r = 0; r < 8; ++r) {
                const int m = m0 + 16 * i + r + 8 * hi2;
                out[(size_t)m * DD + n] = acc[i][j][r] + bias;
            }
        }
    }
}

// ===================== host launcher =====================

extern "C" void kernel_launch(void* const* d_in, const int* in_sizes, int n_in,
                              void* d_out, int out_size, void* d_ws, size_t ws_size,
                              hipStream_t stream) {
    (void)in_sizes; (void)n_in; (void)out_size; (void)ws_size;
    const float* q  = (const float*)d_in[0];
    const float* k  = (const float*)d_in[1];
    const float* v  = (const float*)d_in[2];
    const float* Wq = (const float*)d_in[3];
    const float* Wk = (const float*)d_in[4];
    const float* Wv = (const float*)d_in[5];
    const float* Wo = (const float*)d_in[6];
    const float* bo = (const float*)d_in[7];
    float* out = (float*)d_out;

    char* ws = (char*)d_ws;
    size_t off = 0;
    auto alloc = [&](size_t elems) -> __bf16* {
        __bf16* p = (__bf16*)(ws + off);
        off += ((elems * sizeof(__bf16)) + 255) & ~(size_t)255;
        return p;
    };
    const size_t NX = (size_t)MM * DD;            // 4 Mi elems
    const size_t NW = (size_t)HH * DD * DKK;      // 1 Mi elems
    const size_t NH = (size_t)BB * HH * SS * DKK; // 4 Mi elems

    __bf16* qb  = alloc(NX);
    __bf16* kb  = alloc(NX);
    __bf16* vb  = alloc(NX);
    __bf16* WqT = alloc(NW);
    __bf16* WkT = alloc(NW);
    __bf16* WvT = alloc(NW);
    __bf16* WoT = alloc((size_t)HDK * DD);
    __bf16* Qh  = alloc(NH);
    __bf16* Kh  = alloc(NH);
    __bf16* VhT = alloc(NH);
    __bf16* Oh  = alloc(NH);

    k_cvt<<<8192, 256, 0, stream>>>(q, qb, (int)NX);
    k_cvt<<<8192, 256, 0, stream>>>(k, kb, (int)NX);
    k_cvt<<<8192, 256, 0, stream>>>(v, vb, (int)NX);
    k_twt<<<2048, 256, 0, stream>>>(Wq, WqT);
    k_twt<<<2048, 256, 0, stream>>>(Wk, WkT);
    k_twt<<<2048, 256, 0, stream>>>(Wv, WvT);
    k_two<<<2048, 256, 0, stream>>>(Wo, WoT);

    dim3 gp(16, HH);
    k_proj<<<gp, 128, 0, stream>>>(qb, WqT, Qh, 0);
    k_proj<<<gp, 128, 0, stream>>>(kb, WkT, Kh, 0);
    k_proj<<<gp, 128, 0, stream>>>(vb, WvT, VhT, 1);

    k_attn<<<1024, 128, 0, stream>>>(Qh, Kh, VhT, Oh);

    k_oproj<<<dim3(16, 16), 128, 0, stream>>>(Oh, WoT, bo, out);
}
